// BDHRecurrentAttention_70308614636272
// MI455X (gfx1250) — matrix operand order repaired and checked by reading, not compiled, not run on hardware
//
#include <hip/hip_runtime.h>
#include <hip/hip_bf16.h>

typedef __attribute__((ext_vector_type(16))) __bf16          v16bf;
typedef __attribute__((ext_vector_type(16))) unsigned short  v16u;
typedef __attribute__((ext_vector_type(8)))  unsigned short  u16x8;
typedef __attribute__((ext_vector_type(4)))  unsigned short  u16x4;
typedef __attribute__((ext_vector_type(8)))  float           v8f;

#define B_    2
#define NH_   16
#define T_    2048
#define N_    256
#define D_    1024
#define CHK_  128
#define NCH_  16
#define DT_   64          // D-tile per block

// LDS strides (elements); all multiples of 8 ushorts so rows are 16B-aligned
#define KS_   264         // K tile       [128 rows][256 k]   bf16
#define SS_   136         // scores       [128 rows][128 j]   bf16
#define VTS_  136         // V^T tile     [64 d][128 c]       bf16
#define KTS_  136         // K^T tile     [256 n][128 c]      bf16
#define SBS_  264         // state^T bf16 [64 d][256 n]       bf16

static __device__ __forceinline__ unsigned short f32_to_bf16(float f) {
    return __builtin_bit_cast(unsigned short, (__bf16)f);   // hw cvt, RNE
}

static __device__ __forceinline__ v16bf make_frag(u16x8 lo, u16x8 hi) {
    v16u u = __builtin_shufflevector(lo, hi, 0,1,2,3,4,5,6,7,8,9,10,11,12,13,14,15);
    return __builtin_bit_cast(v16bf, u);
}

// A-fragment (16x32): row-major src[m][k]; two aligned 16B groups per lane
static __device__ __forceinline__ v16bf load_a16(const unsigned short* __restrict__ src,
                                                 int stride, int m0, int k0, int lane) {
    const unsigned short* p = src + (m0 + (lane & 15)) * stride + k0 + ((lane & 16) ? 8 : 0);
    u16x8 lo = *(const u16x8*)p;
    u16x8 hi = *(const u16x8*)(p + 16);
    return make_frag(lo, hi);
}

// B-fragment (32x16): src laid out [n][k] (transposed); two aligned 16B groups per lane
static __device__ __forceinline__ v16bf load_b16(const unsigned short* __restrict__ src,
                                                 int stride, int n0, int k0, int lane) {
    const unsigned short* p = src + (n0 + (lane & 15)) * stride + k0 + ((lane & 16) ? 8 : 0);
    u16x8 lo = *(const u16x8*)p;
    u16x8 hi = *(const u16x8*)(p + 16);
    return make_frag(lo, hi);
}

#define WMMA_BF16(a, b, c) \
    __builtin_amdgcn_wmma_f32_16x16x32_bf16(false, (a), false, (b), (short)0, (c), false, false)

__global__ void __launch_bounds__(256, 1)
bdh_recurrent_attn_kernel(const float* __restrict__ Q,
                          const float* __restrict__ V,
                          float* __restrict__ out) {
    extern __shared__ char smem[];
    unsigned short* Klds  = (unsigned short*)smem;        // [128][KS_]
    unsigned short* Slds  = Klds  + 128 * KS_;            // [128][SS_]
    unsigned short* Vt    = Slds  + 128 * SS_;            // [64][VTS_]
    unsigned short* Kt    = Vt    + 64  * VTS_;           // [256][KTS_]
    unsigned short* StBfT = Kt    + 256 * KTS_;           // [64][SBS_]

    const int tid  = threadIdx.x;
    const int lane = tid & 31;
    const int wv   = tid >> 5;          // wave id 0..7

    const int bid = blockIdx.x;
    const int dt  = bid & 15;           // d-tile
    const int h   = (bid >> 4) & 15;
    const int b   = bid >> 8;
    const int d0  = dt * DT_;

    // persistent fp32 state tiles in WMMA accumulators:
    // wave wv owns n-tile rows {2wv, 2wv+1} x 4 d-tiles
    v8f stacc[2][4];
#pragma unroll
    for (int q2 = 0; q2 < 2; ++q2)
#pragma unroll
        for (int tc = 0; tc < 4; ++tc) stacc[q2][tc] = (v8f)0.0f;

    // zero bf16 transposed state copy (incl. padding)
    for (int i = tid; i < DT_ * SBS_; i += 256) StBfT[i] = 0;
    __syncthreads();

    const float* Qbh0 = Q + ((size_t)(b * NH_ + h) * T_) * N_;
    const float* Vb0  = V + ((size_t)b * T_) * D_ + d0;
    float*       Obh  = out + ((size_t)(b * NH_ + h) * T_) * D_ + d0;

    for (int c = 0; c < NCH_; ++c) {
        // ---- load + RoPE Q-chunk (2 pairs / float4) -> bf16 K + K^T; V^T ----
        const float* Qc = Qbh0 + (size_t)(c * CHK_) * N_;
#pragma unroll 4
        for (int i = 0; i < 32; ++i) {
            int g4  = i * 256 + tid;         // float4 index 0..8191
            int row = g4 >> 6;               // 64 float4 per row
            int p4  = g4 & 63;               // float4 within row
            float4 qv = *(const float4*)(Qc + row * N_ + p4 * 4);
            float t = (float)(c * CHK_ + row);
            // pair frequencies: f(pidx) = 2^(-pidx/8) / (2*pi); pidx = 2*p4, 2*p4+1
            float f0 = exp2f(-(float)(p4 * 2) * 0.125f) * 0.15915494309189535f;
            float f1 = f0 * 0.9170040432046712f;       // * 2^(-1/8)
            float x0 = t * f0; x0 -= floorf(x0);
            float x1 = t * f1; x1 -= floorf(x1);
            float s0, c0, s1, c1;
            __sincosf(x0 * 6.283185307179586f, &s0, &c0);
            __sincosf(x1 * 6.283185307179586f, &s1, &c1);
            u16x4 pk;
            pk.x = f32_to_bf16(qv.x * c0 - qv.y * s0);
            pk.y = f32_to_bf16(qv.y * c0 + qv.x * s0);
            pk.z = f32_to_bf16(qv.z * c1 - qv.w * s1);
            pk.w = f32_to_bf16(qv.w * c1 + qv.z * s1);
            int n = p4 * 4;
            *(u16x4*)(Klds + row * KS_ + n) = pk;      // packed b64 store
            Kt[(n + 0) * KTS_ + row] = pk.x;           // scattered transpose
            Kt[(n + 1) * KTS_ + row] = pk.y;
            Kt[(n + 2) * KTS_ + row] = pk.z;
            Kt[(n + 3) * KTS_ + row] = pk.w;
        }
#pragma unroll 4
        for (int i = 0; i < 8; ++i) {
            int idx  = i * 256 + tid;        // float4 index 0..2047
            int row  = idx >> 4;             // 16 float4 per row slice
            int col4 = (idx & 15) * 4;
            float4 vv = *(const float4*)(Vb0 + (size_t)(c * CHK_ + row) * D_ + col4);
            Vt[(col4 + 0) * VTS_ + row] = f32_to_bf16(vv.x);
            Vt[(col4 + 1) * VTS_ + row] = f32_to_bf16(vv.y);
            Vt[(col4 + 2) * VTS_ + row] = f32_to_bf16(vv.z);
            Vt[(col4 + 3) * VTS_ + row] = f32_to_bf16(vv.w);
        }
        __syncthreads();

        // prefetch next chunk's Q and V slice into cache while we compute
        if (c + 1 < NCH_) {
            const char* Qn = (const char*)(Qbh0 + (size_t)((c + 1) * CHK_) * N_);
#pragma unroll
            for (int i = 0; i < 4; ++i)
                __builtin_prefetch(Qn + (size_t)(i * 256 + tid) * 128, 0, 3);
            if (tid < 128) {
                const char* Vn = (const char*)(Vb0 + (size_t)((c + 1) * CHK_ + tid) * D_);
                __builtin_prefetch(Vn, 0, 3);
            }
        }

        // ---- out_inter = q @ state ; scores = k @ k^T (shared A-fragment) ----
        v8f acc[4];
        v8f sacc[8];
#pragma unroll
        for (int i = 0; i < 4; ++i) acc[i] = (v8f)0.0f;
#pragma unroll
        for (int i = 0; i < 8; ++i) sacc[i] = (v8f)0.0f;

        const int m0 = wv * 16;
#pragma unroll
        for (int ks = 0; ks < 8; ++ks) {
            int k0 = ks * 32;
            v16bf a = load_a16(Klds, KS_, m0, k0, lane);
#pragma unroll
            for (int tc = 0; tc < 4; ++tc)
                acc[tc] = WMMA_BF16(a, load_b16(StBfT, SBS_, tc * 16, k0, lane), acc[tc]);
#pragma unroll
            for (int jt = 0; jt < 8; ++jt)
                sacc[jt] = WMMA_BF16(a, load_b16(Klds, KS_, jt * 16, k0, lane), sacc[jt]);
        }

        // mask (strictly lower-triangular) and write scores to LDS as bf16
#pragma unroll
        for (int jt = 0; jt < 8; ++jt) {
#pragma unroll
            for (int r = 0; r < 8; ++r) {
                int i = m0 + r + ((lane & 16) ? 8 : 0);
                int j = jt * 16 + (lane & 15);
                float sv = (i > j) ? sacc[jt][r] : 0.0f;
                Slds[i * SS_ + j] = f32_to_bf16(sv);
            }
        }
        __syncthreads();   // scores visible; all old-state (StBfT) reads done

        // ---- out += scores @ v ----
#pragma unroll
        for (int ks = 0; ks < 4; ++ks) {
            int k0 = ks * 32;
            v16bf a = load_a16(Slds, SS_, m0, k0, lane);
#pragma unroll
            for (int tc = 0; tc < 4; ++tc)
                acc[tc] = WMMA_BF16(a, load_b16(Vt, VTS_, tc * 16, k0, lane), acc[tc]);
        }

        // store fp32 output tile
#pragma unroll
        for (int tc = 0; tc < 4; ++tc) {
#pragma unroll
            for (int r = 0; r < 8; ++r) {
                int row = m0 + r + ((lane & 16) ? 8 : 0);
                int dd  = tc * 16 + (lane & 15);
                Obh[(size_t)(c * CHK_ + row) * D_ + dd] = acc[tc][r];
            }
        }

        // ---- state += k^T @ v : accumulate directly in WMMA registers,
        //      then refresh the bf16 transposed LDS copy ----
#pragma unroll
        for (int q2 = 0; q2 < 2; ++q2) {
            int nt = wv * 2 + q2;
#pragma unroll
            for (int ks = 0; ks < 4; ++ks) {
                int k0 = ks * 32;
                v16bf a = load_a16(Kt, KTS_, nt * 16, k0, lane);
#pragma unroll
                for (int tc = 0; tc < 4; ++tc)
                    stacc[q2][tc] = WMMA_BF16(a, load_b16(Vt, VTS_, tc * 16, k0, lane),
                                              stacc[q2][tc]);
            }
#pragma unroll
            for (int tc = 0; tc < 4; ++tc) {
#pragma unroll
                for (int r = 0; r < 8; ++r) {
                    int nrow = nt * 16 + r + ((lane & 16) ? 8 : 0);
                    int col  = tc * 16 + (lane & 15);
                    StBfT[col * SBS_ + nrow] = f32_to_bf16(stacc[q2][tc][r]);
                }
            }
        }
        __syncthreads();   // state copy updated & tiles free before next chunk
    }
}

extern "C" void kernel_launch(void* const* d_in, const int* in_sizes, int n_in,
                              void* d_out, int out_size, void* d_ws, size_t ws_size,
                              hipStream_t stream) {
    (void)in_sizes; (void)n_in; (void)d_ws; (void)ws_size; (void)out_size;
    const float* Q = (const float*)d_in[0];
    const float* V = (const float*)d_in[1];
    float* out     = (float*)d_out;

    const size_t shmem =
        (size_t)(128 * KS_ + 128 * SS_ + 64 * VTS_ + 256 * KTS_ + 64 * SBS_)
        * sizeof(unsigned short);   // 223,232 bytes

    dim3 grid(B_ * NH_ * (D_ / DT_));   // 512 blocks
    dim3 block(256);                    // 8 waves (wave32)
    bdh_recurrent_attn_kernel<<<grid, block, shmem, stream>>>(Q, V, out);
}
